// DSSM_66546223284828
// MI455X (gfx1250) — compile-verified, matrix operand order checked by reading, not run on hardware
//
#include <hip/hip_runtime.h>
#include <hip/hip_bf16.h>

// ---------------------------------------------------------------------------
// Problem constants (from reference)
// ---------------------------------------------------------------------------
#define DIMV   1024
#define EDV    2048
#define TWO_ED 4096
#define SNV    16      // SSM state dim
#define BSZ    4
#define LV     4096
#define MTOT   (BSZ * LV)   // 16384 rows

typedef unsigned short u16;
typedef unsigned int   u32;

typedef __attribute__((ext_vector_type(16))) __bf16 v16bf;
typedef __attribute__((ext_vector_type(8)))  float  v8f;
typedef __attribute__((ext_vector_type(4)))  u32    u32x4;
typedef __attribute__((ext_vector_type(4)))  float  f32x4;

union FragBF16 {
    v16bf v;
    u32x4 q[2];
};

__device__ __forceinline__ u16 f2bf(float f) {
    u32 u = __float_as_uint(f);
    u32 r = u + 0x7FFFu + ((u >> 16) & 1u);   // round-to-nearest-even
    return (u16)(r >> 16);
}

// ---------------------------------------------------------------------------
// fp32 -> bf16 bulk convert (8 floats / thread, 16B stores)
// ---------------------------------------------------------------------------
__global__ __launch_bounds__(256)
void cvt_bf16x8(const float* __restrict__ in, u16* __restrict__ out, long n8) {
    long i = (long)blockIdx.x * blockDim.x + threadIdx.x;
    if (i >= n8) return;
    const f32x4* in4 = (const f32x4*)in;
    f32x4 a = in4[2 * i];
    f32x4 b = in4[2 * i + 1];
    u32x4 o;
    o[0] = (u32)f2bf(a[0]) | ((u32)f2bf(a[1]) << 16);
    o[1] = (u32)f2bf(a[2]) | ((u32)f2bf(a[3]) << 16);
    o[2] = (u32)f2bf(b[0]) | ((u32)f2bf(b[1]) << 16);
    o[3] = (u32)f2bf(b[2]) | ((u32)f2bf(b[3]) << 16);
    ((u32x4*)out)[i] = o;
}

// ---------------------------------------------------------------------------
// WMMA bf16 GEMM:  C[M,N] = A[M,K](bf16) * B[N,K](bf16)^T  (+ Res)
// 8 waves/WG arranged 2 (M) x 4 (N); each wave computes a 64x64 tile
// (4x4 WMMA 16x16x32 tiles) -> 32 FLOP/byte fragment-load intensity.
// WG tile: 128 (M) x 256 (N).  A/B fragments loaded directly from row-major
// bf16 per the CDNA5 16-bit A/B VGPR layout (lane%16 = row, lane/16 selects
// the +8 K-offset half; elems 0-7 -> K block k0.., elems 8-15 -> k0+16..).
// ---------------------------------------------------------------------------
__global__ __launch_bounds__(256)
void gemm_bf16_wmma(const u16* __restrict__ A, const u16* __restrict__ Bw,
                    const float* __restrict__ Res, float* __restrict__ C,
                    int M, int Nn, int K)
{
    const int lane = threadIdx.x & 31;
    const int w    = threadIdx.x >> 5;      // 0..7
    const int wm   = w >> 2;                // 0..1
    const int wn   = w & 3;                 // 0..3
    const int m_base = blockIdx.y * 128 + wm * 64;
    const int n_base = blockIdx.x * 256 + wn * 64;
    const int lr = lane & 15;
    const int hh = lane >> 4;

    const v8f zero = {0.f, 0.f, 0.f, 0.f, 0.f, 0.f, 0.f, 0.f};
    v8f acc[4][4];
#pragma unroll
    for (int i = 0; i < 4; i++)
#pragma unroll
        for (int j = 0; j < 4; j++) acc[i][j] = zero;

    const u16* Arow[4];
    const u16* Brow[4];
#pragma unroll
    for (int i = 0; i < 4; i++) {
        Arow[i] = A  + (size_t)(m_base + 16 * i + lr) * K;
        Brow[i] = Bw + (size_t)(n_base + 16 * i + lr) * K;
    }

#pragma unroll 1
    for (int k0 = 0; k0 < K; k0 += 32) {
        if (k0 + 32 < K) {   // emits global_prefetch_b8 on gfx1250
            __builtin_prefetch(Arow[0] + k0 + 32, 0, 1);
            __builtin_prefetch(Arow[2] + k0 + 32, 0, 1);
            __builtin_prefetch(Brow[0] + k0 + 32, 0, 1);
            __builtin_prefetch(Brow[2] + k0 + 32, 0, 1);
        }
        const int ka = k0 + hh * 8;
        FragBF16 af[4], bfm[4];
#pragma unroll
        for (int i = 0; i < 4; i++) {
            af[i].q[0]  = *(const u32x4*)(Arow[i] + ka);
            af[i].q[1]  = *(const u32x4*)(Arow[i] + ka + 16);
            bfm[i].q[0] = *(const u32x4*)(Brow[i] + ka);
            bfm[i].q[1] = *(const u32x4*)(Brow[i] + ka + 16);
        }
#pragma unroll
        for (int i = 0; i < 4; i++)
#pragma unroll
            for (int j = 0; j < 4; j++)
                acc[i][j] = __builtin_amdgcn_wmma_f32_16x16x32_bf16(
                    false, af[i].v, false, bfm[j].v,
                    (short)0, acc[i][j], false, false);
    }

    // D layout: VGPR r -> row (r + 8*hh), lane%16 -> col
#pragma unroll
    for (int i = 0; i < 4; i++) {
        const int mrow0 = m_base + i * 16 + hh * 8;
#pragma unroll
        for (int j = 0; j < 4; j++) {
            const int col = n_base + j * 16 + lr;
#pragma unroll
            for (int r = 0; r < 8; r++) {
                size_t idx = (size_t)(mrow0 + r) * Nn + col;
                float v = acc[i][j][r];
                if (Res) v += Res[idx];
                C[idx] = v;
            }
        }
    }
}

// ---------------------------------------------------------------------------
// P[n] = sum_e W_dt[e] * A[e][n]   (a[b,l,n] = dt[b,l] * P[n])
// ---------------------------------------------------------------------------
__global__ void precompute_P(const float* __restrict__ A,
                             const float* __restrict__ W_dt,
                             float* __restrict__ P)
{
    int n = threadIdx.x;
    if (n >= SNV) return;
    float s = 0.f;
    for (int e = 0; e < EDV; e++) s = fmaf(W_dt[e], A[e * SNV + n], s);
    P[n] = s;
}

// ---------------------------------------------------------------------------
// Per (b,l): dt = mean(x_ssm); u[n] = dt * sum_e W_dt[e]*x_ssm[e]*Bm[e][n];
//            a[n] = dt * P[n].   One 256-thread block per row, 8 chans/thread.
// ---------------------------------------------------------------------------
__global__ __launch_bounds__(256)
void dtau_kernel(const float* __restrict__ proj, const float* __restrict__ Bm,
                 const float* __restrict__ W_dt, const float* __restrict__ P,
                 float* __restrict__ a_out, float* __restrict__ u_out)
{
    __shared__ float red[SNV * 256];
    const int row = blockIdx.x;
    const int tid = threadIdx.x;
    const int e0  = tid * 8;
    const float* xrow = proj + (size_t)row * TWO_ED + EDV;

    float xs[8], wd[8];
    float s = 0.f;
#pragma unroll
    for (int j = 0; j < 8; j++) {
        xs[j] = xrow[e0 + j];
        s += xs[j];
        wd[j] = W_dt[e0 + j] * xs[j];
    }
    red[tid] = s;
    __syncthreads();
    for (int st = 128; st >= 1; st >>= 1) {
        if (tid < st) red[tid] += red[tid + st];
        __syncthreads();
    }
    const float dtv = red[0] * (1.0f / (float)EDV);
    __syncthreads();

    float un[SNV];
#pragma unroll
    for (int n = 0; n < SNV; n++) un[n] = 0.f;
#pragma unroll
    for (int j = 0; j < 8; j++) {
        const float* brow = Bm + (size_t)(e0 + j) * SNV;
#pragma unroll
        for (int n = 0; n < SNV; n++) un[n] = fmaf(wd[j], brow[n], un[n]);
    }
#pragma unroll
    for (int n = 0; n < SNV; n++) red[n * 256 + tid] = un[n];
    __syncthreads();
    for (int st = 128; st >= 1; st >>= 1) {
        if (tid < st)
#pragma unroll
            for (int n = 0; n < SNV; n++)
                red[n * 256 + tid] += red[n * 256 + tid + st];
        __syncthreads();
    }
    if (tid < SNV) {
        u_out[(size_t)row * SNV + tid] = dtv * red[tid * 256];
        a_out[(size_t)row * SNV + tid] = dtv * P[tid];
    }
}

// ---------------------------------------------------------------------------
// Parallel linear scan m_t = a_t*m_{t-1} + u_t over L=4096, per (b,n).
// 32 chunks of 128.  t = ((b*16+n)*32 + c).
// ---------------------------------------------------------------------------
#define NCHUNK 32
#define CLEN   128

__global__ void scan_phase1(const float* __restrict__ a, const float* __restrict__ u,
                            float* __restrict__ Ac, float* __restrict__ Mc)
{
    int t = blockIdx.x * blockDim.x + threadIdx.x;
    if (t >= BSZ * SNV * NCHUNK) return;
    int c = t & (NCHUNK - 1);
    int n = (t >> 5) & (SNV - 1);
    int b = t >> 9;
    size_t base = ((size_t)b * LV + (size_t)c * CLEN) * SNV + n;
    float Ap = 1.f, m = 0.f;
    for (int i = 0; i < CLEN; i++) {
        float at = a[base + (size_t)i * SNV];
        float ut = u[base + (size_t)i * SNV];
        Ap *= at;
        m = fmaf(at, m, ut);
    }
    Ac[t] = Ap;
    Mc[t] = m;
}

__global__ void scan_phase2(const float* __restrict__ Ac, const float* __restrict__ Mc,
                            float* __restrict__ Minit)
{
    int t = threadIdx.x;                    // 0..63  ->  (b,n)
    if (t >= BSZ * SNV) return;
    int base = t * NCHUNK;
    float m = 0.f;
    for (int c = 0; c < NCHUNK; c++) {
        Minit[base + c] = m;
        m = fmaf(Ac[base + c], m, Mc[base + c]);
    }
}

__global__ void scan_phase3(const float* __restrict__ a, const float* __restrict__ u,
                            const float* __restrict__ Minit, float* __restrict__ mseq)
{
    int t = blockIdx.x * blockDim.x + threadIdx.x;
    if (t >= BSZ * SNV * NCHUNK) return;
    int c = t & (NCHUNK - 1);
    int n = (t >> 5) & (SNV - 1);
    int b = t >> 9;
    size_t base = ((size_t)b * LV + (size_t)c * CLEN) * SNV + n;
    float m = Minit[t];
    for (int i = 0; i < CLEN; i++) {
        size_t idx = base + (size_t)i * SNV;
        m = fmaf(a[idx], m, u[idx]);
        mseq[idx] = m;
    }
}

// ---------------------------------------------------------------------------
// Fusion: y = mseq@Cm + Dv*x_ssm ; x_conv = depthwise conv3 of proj[:, :ED];
// gate = sigmoid(y); z = y + gate*(x_conv - y); store z as bf16.
// One block per (b,l), 8 channels per thread.
// ---------------------------------------------------------------------------
__global__ __launch_bounds__(256)
void fuse_kernel(const float* __restrict__ proj, const float* __restrict__ mseq,
                 const float* __restrict__ Cm, const float* __restrict__ Dv,
                 const float* __restrict__ conv_w, u16* __restrict__ zb)
{
    __shared__ float ms[SNV];
    const int row = blockIdx.x;
    const int l   = row & (LV - 1);
    const int tid = threadIdx.x;
    const int e0  = tid * 8;

    if (tid < SNV) ms[tid] = mseq[(size_t)row * SNV + tid];
    __syncthreads();

    const float* prow = proj + (size_t)row * TWO_ED;

    float y[8];
#pragma unroll
    for (int j = 0; j < 8; j++) y[j] = Dv[e0 + j] * prow[EDV + e0 + j];
#pragma unroll
    for (int n = 0; n < SNV; n++) {
        const float mn = ms[n];
        const float* crow = Cm + (size_t)n * EDV + e0;
#pragma unroll
        for (int j = 0; j < 8; j++) y[j] = fmaf(mn, crow[j], y[j]);
    }

    float xc[8];
#pragma unroll
    for (int j = 0; j < 8; j++) xc[j] = 0.f;
    for (int kk = 0; kk < 3; kk++) {
        int ll = l + kk - 1;
        if (ll < 0 || ll >= LV) continue;
        const float* pc = proj + (size_t)(row + kk - 1) * TWO_ED + e0;
#pragma unroll
        for (int j = 0; j < 8; j++)
            xc[j] = fmaf(conv_w[(size_t)(e0 + j) * 3 + kk], pc[j], xc[j]);
    }

    u16 h[8];
#pragma unroll
    for (int j = 0; j < 8; j++) {
        float g = 1.f / (1.f + __expf(-y[j]));
        float z = fmaf(g, xc[j] - y[j], y[j]);   // xc*g + y*(1-g)
        h[j] = f2bf(z);
    }
    u32x4 o;
    o[0] = (u32)h[0] | ((u32)h[1] << 16);
    o[1] = (u32)h[2] | ((u32)h[3] << 16);
    o[2] = (u32)h[4] | ((u32)h[5] << 16);
    o[3] = (u32)h[6] | ((u32)h[7] << 16);
    ((u32x4*)(zb + (size_t)row * EDV))[tid] = o;
}

// ---------------------------------------------------------------------------
// Host launcher
// ---------------------------------------------------------------------------
extern "C" void kernel_launch(void* const* d_in, const int* in_sizes, int n_in,
                              void* d_out, int out_size, void* d_ws, size_t ws_size,
                              hipStream_t stream)
{
    const float* x      = (const float*)d_in[0];   // [4,4096,1024]
    const float* Amat   = (const float*)d_in[1];   // [2048,16]
    const float* Bm     = (const float*)d_in[2];   // [2048,16]
    const float* Cm     = (const float*)d_in[3];   // [16,2048]
    const float* Dv     = (const float*)d_in[4];   // [2048]
    const float* W_dt   = (const float*)d_in[5];   // [2048,1]
    const float* conv_w = (const float*)d_in[6];   // [2048,1,3]
    const float* W_in   = (const float*)d_in[7];   // [4096,1024]
    const float* W_out  = (const float*)d_in[8];   // [1024,2048]
    float* out = (float*)d_out;                    // [4,4096,1024]

    char* ws = (char*)d_ws;
    size_t off = 0;
    auto take = [&](size_t bytes) -> char* {
        char* p = ws + off;
        off = (off + bytes + 255) & ~(size_t)255;
        return p;
    };
    u16*   xb    = (u16*)  take((size_t)MTOT * DIMV * 2);      // 32 MB
    u16*   Wib   = (u16*)  take((size_t)TWO_ED * DIMV * 2);    //  8 MB
    u16*   Wob   = (u16*)  take((size_t)DIMV * EDV * 2);       //  4 MB
    float* proj  = (float*)take((size_t)MTOT * TWO_ED * 4);    // 256 MB
    u16*   zb    = (u16*)  take((size_t)MTOT * EDV * 2);       // 64 MB
    float* a_buf = (float*)take((size_t)MTOT * SNV * 4);       //  1 MB
    float* u_buf = (float*)take((size_t)MTOT * SNV * 4);       //  1 MB
    float* mseq  = (float*)take((size_t)MTOT * SNV * 4);       //  1 MB
    float* Ac    = (float*)take((size_t)BSZ * SNV * NCHUNK * 4);
    float* Mc    = (float*)take((size_t)BSZ * SNV * NCHUNK * 4);
    float* Minit = (float*)take((size_t)BSZ * SNV * NCHUNK * 4);
    float* Pbuf  = (float*)take(SNV * 4);

    // 1) bf16 conversions
    {
        long n8 = (long)MTOT * DIMV / 8;
        cvt_bf16x8<<<(unsigned)(n8 / 256), 256, 0, stream>>>(x, xb, n8);
        n8 = (long)TWO_ED * DIMV / 8;
        cvt_bf16x8<<<(unsigned)(n8 / 256), 256, 0, stream>>>(W_in, Wib, n8);
        n8 = (long)DIMV * EDV / 8;
        cvt_bf16x8<<<(unsigned)(n8 / 256), 256, 0, stream>>>(W_out, Wob, n8);
    }

    // 2) P[n] = sum_e W_dt[e]*A[e][n]
    precompute_P<<<1, 32, 0, stream>>>(Amat, W_dt, Pbuf);

    // 3) in_proj GEMM: proj[16384,4096] = xb * Wib^T   (WMMA bf16)
    {
        dim3 grid(TWO_ED / 256, MTOT / 128);
        gemm_bf16_wmma<<<grid, 256, 0, stream>>>(xb, Wib, nullptr, proj,
                                                 MTOT, TWO_ED, DIMV);
    }

    // 4) dt / a / u
    dtau_kernel<<<MTOT, 256, 0, stream>>>(proj, Bm, W_dt, Pbuf, a_buf, u_buf);

    // 5) parallel scan
    scan_phase1<<<(BSZ * SNV * NCHUNK) / 256, 256, 0, stream>>>(a_buf, u_buf, Ac, Mc);
    scan_phase2<<<1, 64, 0, stream>>>(Ac, Mc, Minit);
    scan_phase3<<<(BSZ * SNV * NCHUNK) / 256, 256, 0, stream>>>(a_buf, u_buf, Minit, mseq);

    // 6) y / conv / gate fusion -> z (bf16)
    fuse_kernel<<<MTOT, 256, 0, stream>>>(proj, mseq, Cm, Dv, conv_w, zb);

    // 7) out_proj GEMM + residual: out[16384,1024] = zb * Wob^T + x
    {
        dim3 grid(DIMV / 256, MTOT / 128);
        gemm_bf16_wmma<<<grid, 256, 0, stream>>>(zb, Wob, x, out,
                                                 MTOT, DIMV, EDV);
    }
    (void)in_sizes; (void)n_in; (void)out_size; (void)ws_size;
}
